// GNNDecoder_14482629722139
// MI455X (gfx1250) — matrix operand order repaired and checked by reading, not compiled, not run on hardware
//
#include <hip/hip_runtime.h>
#include <hip/hip_bf16.h>

#define N_NODES  50216
#define NUM_DRUG 216
#define NUM_RNA  (N_NODES - NUM_DRUG)   // 50000 = 3125 * 16 (exact M tiles)
#define N_EDGES  803456
#define D_IN     64
#define D_HID    64
#define D_OUT    32
#define N_PAD    50224                  // N_NODES rounded up to 16
#define N_TILES  (N_PAD / 16)           // 3139

typedef __attribute__((ext_vector_type(16))) _Float16 v16h;
typedef __attribute__((ext_vector_type(8)))  float    v8f;

// ---------------------------------------------------------------------------
// WMMA A-fragment loader (gfx1250 wave32; contiguous k order, same as packed B frags)
// 16x32 f16 A tile from a row-major f32 source (row stride ld), rows
// [row0,row0+16), cols [k0,k0+32), scaled by `scale` (folds the 1/max(cnt,1)
// mean-aggregation into the load).  Lane L<16: row L, K = 0..15;
// lane L>=16: row L-16, K = 16..31.
// ---------------------------------------------------------------------------
static __device__ inline v16h load_a_f16(const float* __restrict__ src, int ld,
                                         int row0, int k0, float scale,
                                         int lane, int row_max) {
  int r = lane & 15;
  int kb = (lane >> 4) * 16;
  int row = row0 + r;
  if (row >= row_max) row = row_max - 1;       // only live for unpadded x
  const float* p = src + (size_t)row * ld + k0;
  v16h a;
#pragma unroll
  for (int i = 0; i < 8; ++i) {
    a[i]     = (_Float16)(p[kb + i] * scale);
    a[i + 8] = (_Float16)(p[kb + 8 + i] * scale);
  }
  return a;
}

// ---------------------------------------------------------------------------
// One-shot weight packer: W (dout x din, row-major f32) -> f16 fragments in
// B-matrix (W^T) wave32 layout, stored contiguously per fragment:
//   frag f = (n0/16)*kt + (k0/32);  half index = f*512 + lane*16 + i
//   lane L<16: column n0+L, K = k0..k0+15; lane L>=16: column n0+L-16,
//   K = k0+16..k0+31.
// Dense kernels then load a whole fragment as one aligned 32B v16h.
// ---------------------------------------------------------------------------
__global__ void pack_weights_kernel(const float* __restrict__ W,
                                    _Float16* __restrict__ dst,
                                    int din, int dout) {
  int kt = din >> 5;                       // k tiles
  int total = (dout >> 4) * kt * 512;      // halves
  for (int e = blockIdx.x * blockDim.x + threadIdx.x; e < total;
       e += gridDim.x * blockDim.x) {
    int f    = e >> 9;
    int j    = e & 511;
    int lane = j >> 4;
    int i    = j & 15;
    int n0   = (f / kt) << 4;
    int k0   = (f % kt) << 5;
    int n    = n0 + (lane & 15);
    int k    = k0 + ((lane >> 4) << 4) + i;
    dst[e] = (_Float16)W[(size_t)n * din + k];
  }
}

// ---------------------------------------------------------------------------
// Graph kernels (bandwidth-bound: HBM/L2 gathers + f32 atomics)
// ---------------------------------------------------------------------------
__global__ void zero_kernel(float* __restrict__ p, long long n) {
  long long i = (long long)blockIdx.x * blockDim.x + threadIdx.x;
  long long stride = (long long)gridDim.x * blockDim.x;
  for (; i < n; i += stride) p[i] = 0.0f;
}

__global__ void degree_kernel(const long long* __restrict__ ei,
                              float* __restrict__ cnt) {
  int e = blockIdx.x * blockDim.x + threadIdx.x;
  if (e >= N_EDGES) return;
  atomicAdd(&cnt[(int)ei[N_EDGES + e]], 1.0f);
}

// One thread per (edge, dim): 64 consecutive lanes read one 256B row of
// feat[src] fully coalesced, then global_atomic_add_f32 into L2-resident sum.
__global__ void scatter64_kernel(const long long* __restrict__ ei,
                                 const float* __restrict__ feat,
                                 float* __restrict__ sum) {
  long long idx = (long long)blockIdx.x * blockDim.x + threadIdx.x;
  if (idx >= (long long)N_EDGES * 64) return;
  int e = (int)(idx >> 6);
  int d = (int)(idx & 63);
  int s = (int)ei[e];
  int t = (int)ei[N_EDGES + e];
  atomicAdd(&sum[(size_t)t * 64 + d], feat[(size_t)s * 64 + d]);
}

// ---------------------------------------------------------------------------
// Fused SAGE dense layer over padded buffers:
//   out = act( (summed/max(cnt,1)) @ Wl^T + bl + xin @ Wr^T )
// One wave per 16x16 output tile; no store guards (out has N_PAD rows).
// din = 64 -> 2 K-steps * 2 operands = 4 WMMAs per tile.  EXEC stays all-1s
// through every WMMA (wave-uniform control flow only).
// ---------------------------------------------------------------------------
__global__ void sage_dense_kernel(const float* __restrict__ summed,  // N_PAD x din
                                  const float* __restrict__ cnt,     // N_PAD
                                  const float* __restrict__ xin,     // xin_rows x din
                                  const _Float16* __restrict__ pWl,  // packed B frags
                                  const float* __restrict__ bl,
                                  const _Float16* __restrict__ pWr,  // packed B frags
                                  float* __restrict__ out,           // N_PAD x dout
                                  int din, int dout, int xin_rows, int do_relu) {
  int lane = threadIdx.x & 31;
  int wave = threadIdx.x >> 5;
  int row0 = blockIdx.x * 16;
  int n0 = wave * 16;
  if (n0 >= dout) return;                       // wave-uniform

  int kt = din >> 5;
  const v16h* bl_frags = (const v16h*)pWl;      // frag*32 + lane (v16h units)
  const v16h* br_frags = (const v16h*)pWr;

  // per-lane inverse degree for the A-fragment row this lane carries
  float inv = 1.0f / fmaxf(cnt[row0 + (lane & 15)], 1.0f);

  v8f acc = {};
  for (int k0 = 0; k0 < din; k0 += 32) {
    size_t fb = (size_t)((n0 >> 4) * kt + (k0 >> 5)) * 32 + lane;

    v16h a_agg = load_a_f16(summed, din, row0, k0, inv, lane, N_PAD);
    v16h b_l   = bl_frags[fb];                  // single aligned 32B load
    acc = __builtin_amdgcn_wmma_f32_16x16x32_f16(
        false, a_agg, false, b_l, (short)0, acc, false, false);

    v16h a_x   = load_a_f16(xin, din, row0, k0, 1.0f, lane, xin_rows);
    v16h b_r   = br_frags[fb];
    acc = __builtin_amdgcn_wmma_f32_16x16x32_f16(
        false, a_x, false, b_r, (short)0, acc, false, false);
  }

  // C/D layout: element i -> row row0 + (lane>>4)*8 + i, col n0 + (lane&15)
  int col  = n0 + (lane & 15);
  float bias = bl[col];
  float* po = out + (size_t)(row0 + ((lane >> 4) << 3)) * dout + col;
#pragma unroll
  for (int i = 0; i < 8; ++i) {
    float v = acc[i] + bias;
    if (do_relu) v = fmaxf(v, 0.0f);
    po[(size_t)i * dout] = v;                   // unguarded: padded rows
  }
}

// ---------------------------------------------------------------------------
// Decoder: out[m][n] = sum_k rna[m][k] * drug[n][k], K = 32 -> exactly one
// v_wmma_f32_16x16x32_f16 per 16x16 tile.  M = 50000 is tile-exact; only the
// ragged drug-column edge needs a (hoisted, i-independent) store guard.
// h2 is padded to N_PAD rows so even n up to 223 reads in-bounds garbage.
// ---------------------------------------------------------------------------
__global__ void decoder_kernel(const float* __restrict__ h2,
                               float* __restrict__ out) {
  int lane = threadIdx.x & 31;
  int m0 = blockIdx.x * 16;
  int n0 = blockIdx.y * 16;

  // A fragment: rna rows m0..m0+15 (< 50000 always), full K = 32
  const float* pa = h2 + (size_t)(m0 + (lane & 15)) * D_OUT;
  int kb = (lane >> 4) * 16;
  v16h a;
#pragma unroll
  for (int i = 0; i < 8; ++i) {
    a[i]     = (_Float16)pa[kb + i];
    a[i + 8] = (_Float16)pa[kb + 8 + i];
  }

  // B fragment: B[k][n] = h2[(NUM_RNA + n)*32 + k]; rows stay < N_PAD
  int ncol = n0 + (lane & 15);
  const float* pb = h2 + (size_t)(NUM_RNA + ncol) * D_OUT + ((lane >> 4) << 4);
  v16h b;
#pragma unroll
  for (int i = 0; i < 16; ++i) b[i] = (_Float16)pb[i];

  v8f acc = {};
  acc = __builtin_amdgcn_wmma_f32_16x16x32_f16(
      false, a, false, b, (short)0, acc, false, false);

  if (ncol < NUM_DRUG) {                        // single EXEC mask, 8 stores
    float* po = out + (size_t)(m0 + ((lane >> 4) << 3)) * NUM_DRUG + ncol;
#pragma unroll
    for (int i = 0; i < 8; ++i) po[(size_t)i * NUM_DRUG] = acc[i];
  }
}

// ---------------------------------------------------------------------------
// Launch
// ---------------------------------------------------------------------------
extern "C" void kernel_launch(void* const* d_in, const int* in_sizes, int n_in,
                              void* d_out, int out_size, void* d_ws, size_t ws_size,
                              hipStream_t stream) {
  const float*     x   = (const float*)d_in[0];
  const long long* ei  = (const long long*)d_in[1];   // (2, N_EDGES) int64
  const float*     Wl1 = (const float*)d_in[2];
  const float*     bl1 = (const float*)d_in[3];
  const float*     Wr1 = (const float*)d_in[4];
  const float*     Wl2 = (const float*)d_in[5];
  const float*     bl2 = (const float*)d_in[6];
  const float*     Wr2 = (const float*)d_in[7];
  float* out = (float*)d_out;

  // Workspace layout (floats first, then 32B-aligned f16 packed weights)
  float* ws      = (float*)d_ws;
  float* summed1 = ws;                                  // N_PAD * 64
  float* cnt     = summed1 + (size_t)N_PAD * 64;        // N_PAD
  float* summed2 = cnt + N_PAD;                         // N_PAD * 64
  float* h1      = summed2 + (size_t)N_PAD * 64;        // N_PAD * 64
  float* h2      = h1 + (size_t)N_PAD * 64;             // N_PAD * 32
  _Float16* pW   = (_Float16*)(h2 + (size_t)N_PAD * 32); // 32B-aligned
  _Float16* pWl1 = pW;                                   // 4 nt * 2 kt * 512
  _Float16* pWr1 = pWl1 + 4096;
  _Float16* pWl2 = pWr1 + 4096;                          // 2 nt * 2 kt * 512
  _Float16* pWr2 = pWl2 + 2048;

  // 0) pack all four weight matrices into B-fragment order (tiny, cached)
  pack_weights_kernel<<<16, 256, 0, stream>>>(Wl1, pWl1, D_IN,  D_HID);
  pack_weights_kernel<<<16, 256, 0, stream>>>(Wr1, pWr1, D_IN,  D_HID);
  pack_weights_kernel<<<16, 256, 0, stream>>>(Wl2, pWl2, D_HID, D_OUT);
  pack_weights_kernel<<<16, 256, 0, stream>>>(Wr2, pWr2, D_HID, D_OUT);

  // 1) zero accumulators (summed1 | cnt | summed2 contiguous, padded)
  zero_kernel<<<2048, 256, 0, stream>>>(summed1, (long long)N_PAD * 129);

  // 2) in-degree counts (shared by both layers)
  degree_kernel<<<(N_EDGES + 255) / 256, 256, 0, stream>>>(ei, cnt);

  // 3) layer-1 message scatter-add
  long long nsc = (long long)N_EDGES * 64;
  int scgrid = (int)((nsc + 255) / 256);
  scatter64_kernel<<<scgrid, 256, 0, stream>>>(ei, x, summed1);

  // 4) layer-1 dense: h1 = relu(agg1 @ Wl1^T + bl1 + x @ Wr1^T)
  sage_dense_kernel<<<N_TILES, 128, 0, stream>>>(summed1, cnt, x,
                                                 pWl1, bl1, pWr1, h1,
                                                 D_IN, D_HID, N_NODES, 1);

  // 5) layer-2 message scatter-add (h1 is padded -> no clamp in dense)
  scatter64_kernel<<<scgrid, 256, 0, stream>>>(ei, h1, summed2);

  // 6) layer-2 dense: h2 = agg2 @ Wl2^T + bl2 + h1 @ Wr2^T
  sage_dense_kernel<<<N_TILES, 64, 0, stream>>>(summed2, cnt, h1,
                                                pWl2, bl2, pWr2, h2,
                                                D_HID, D_OUT, N_PAD, 0);

  // 7) decoder: (50000 x 32) @ (216 x 32)^T, one WMMA per 16x16 tile
  dim3 dg(NUM_RNA / 16, (NUM_DRUG + 15) / 16);
  decoder_kernel<<<dg, 32, 0, stream>>>(h2, out);
}
